// STGCN_88081189306825
// MI455X (gfx1250) — compile-verified, matrix operand order checked by reading, not run on hardware
//
#include <hip/hip_runtime.h>
#include <math.h>

typedef __attribute__((ext_vector_type(16))) _Float16 v16h;
typedef __attribute__((ext_vector_type(8)))  _Float16 v8h;
typedef __attribute__((ext_vector_type(8)))  float    v8f;

#define N_NODES 10000
#define N_FEAT  32
#define HID     64
#define OUTF    16
#define N_EDGES 160000
#define T_SEQ   12
#define NT      (N_NODES * T_SEQ)   // 120000 rows for GCN-batched GEMMs

// ---------------------------------------------------------------------------
// K-index mapping for the 16-bit 16x32 A/B WMMA operand layout (wave32):
// lanes 0-15 hold K 0-7 (elem 0-7) and K 16-23 (elem 8-15); lanes 16-31 hold
// K 8-15 and K 24-31.  row/col = lane & 15.
// ---------------------------------------------------------------------------
__device__ __forceinline__ int kmap(int j, int half) {
  return (j & 7) + (half << 3) + ((j >> 3) << 4);
}

// ---------------------------------------------------------------------------
// LDS-staged f16 WMMA GEMM:
//   C[M, gridDim.y*NCOLS*16] = A @ op(B) (+bias) (ReLU?)
//   BT=false : B is (K,N) row-major (GCN weights)
//   BT=true  : B is (N,K) row-major (LSTM wih/whh, fc_w -> implicit B^T)
//   DUAL     : K=128 split at 64:  A = [A0 | A1],  B^T = [B0 | B1]
//              (fused LSTM gates = [x_t ; h] @ [wih ; whh]^T)
//
// B is staged into LDS *pre-swizzled into WMMA fragment order*: slot
//   idx(chunk, c, jh, lane, e) = (((chunk*NCOLS + c)*2 + jh)*32 + lane)*8 + e
// holds B[chunk*32 + kmap(jh*8+e, lane>>4)][c*16 + (lane&15)] as f16, so the
// consuming lane reads its whole v16h operand as two contiguous
// ds_load_b128 (conflict-free: 16B per lane, linear bank sweep).
// ---------------------------------------------------------------------------
template<int KT, int NCOLS, bool BT, bool DUAL, bool BIAS, bool RELU>
__global__ void wmma_gemm_lds(const float* __restrict__ A0, int lda0,
                              const float* __restrict__ A1, int lda1,
                              const float* __restrict__ B0, int ldb0,
                              const float* __restrict__ B1, int ldb1,
                              float* __restrict__ C, int ldc,
                              const float* __restrict__ bias, int M) {
  constexpr int NC16   = NCOLS * 16;
  constexpr int NCHUNK = KT / 32;
  __shared__ __align__(16) _Float16 bsh[KT * NC16];

  // ---- cooperative B staging: global f32 -> LDS f16 in fragment order ----
  const int nbase = blockIdx.y * NC16;         // block's first output column
  for (int i = threadIdx.x; i < KT * NC16; i += 256) {
    const int e     = i & 7;
    const int lane  = (i >> 3) & 31;
    const int jh    = (i >> 8) & 1;
    const int rest  = i >> 9;
    const int c     = rest % NCOLS;
    const int chunk = rest / NCOLS;
    const int j     = jh * 8 + e;
    const int k     = chunk * 32 + kmap(j, lane >> 4);
    const int gn    = nbase + c * 16 + (lane & 15);
    float v;
    if (BT) {
      if (DUAL) v = (k < 64) ? B0[(long)gn * ldb0 + k]
                             : B1[(long)gn * ldb1 + (k - 64)];
      else      v = B0[(long)gn * ldb0 + k];
    } else {
      v = B0[(long)k * ldb0 + gn];
    }
    bsh[i] = (_Float16)v;
  }
  __syncthreads();

  // ---- per-wave 16 x (NCOLS*16) output strip -----------------------------
  const int mt = blockIdx.x * 8 + (threadIdx.x >> 5);
  if (mt >= (M >> 4)) return;                  // wave-uniform exit
  const int lane = threadIdx.x & 31;
  const int half = lane >> 4;
  const int rc   = lane & 15;
  const int arow = (mt << 4) + rc;

  __builtin_prefetch(A0 + (long)arow * lda0, 0, 3);   // global_prefetch_b8

  v8f acc[NCOLS];
#pragma unroll
  for (int c = 0; c < NCOLS; ++c) acc[c] = {};

#pragma unroll
  for (int chunk = 0; chunk < NCHUNK; ++chunk) {
    const int k0 = chunk * 32;
    const float* Abase;
    int kbase;
    if (!DUAL || k0 < 64) { Abase = A0 + (long)arow * lda0; kbase = k0; }
    else                  { Abase = A1 + (long)arow * lda1; kbase = k0 - 64; }

    // A fragment: 8x global_load_b64 (kmap pairs are K-consecutive) -> f16
    v16h a;
#pragma unroll
    for (int p = 0; p < 8; ++p) {
      const int kk = kbase + ((2 * p) & 7) + (half << 3) + ((p >> 2) << 4);
      const float2 av = *(const float2*)(Abase + kk);
      a[2 * p]     = (_Float16)av.x;
      a[2 * p + 1] = (_Float16)av.y;
    }

#pragma unroll
    for (int c = 0; c < NCOLS; ++c) {
      const int ofs = (((chunk * NCOLS + c) * 2) * 32 + lane) * 8;
      const v8h lo = *(const v8h*)&bsh[ofs];         // ds_load_b128
      const v8h hi = *(const v8h*)&bsh[ofs + 256];   // ds_load_b128 (jh=1)
      const v16h b = __builtin_shufflevector(lo, hi, 0, 1, 2, 3, 4, 5, 6, 7,
                                             8, 9, 10, 11, 12, 13, 14, 15);
      acc[c] = __builtin_amdgcn_wmma_f32_16x16x32_f16(
          false, a, false, b, (short)0, acc[c], false, false);
    }
  }

  const int mbase = (mt << 4) + (half << 3);   // C rows: mbase..mbase+7
#pragma unroll
  for (int c = 0; c < NCOLS; ++c) {
    const int gcol = nbase + c * 16 + rc;
    const float bv = BIAS ? bias[gcol] : 0.0f;
#pragma unroll
    for (int i = 0; i < 8; ++i) {
      float v = acc[c][i] + bv;
      if (RELU) v = fmaxf(v, 0.0f);
      C[(long)(mbase + i) * ldc + gcol] = v;
    }
  }
}

// ---------------------------------------------------------------------------
// Elementwise / graph kernels
// ---------------------------------------------------------------------------
__global__ void transpose_x_kernel(const float* __restrict__ x,
                                   float* __restrict__ xs) {
  // x: (T, N, F)  ->  xs: (N*T, F)
  int idx = blockIdx.x * blockDim.x + threadIdx.x;
  if (idx >= NT * N_FEAT) return;
  int f = idx % N_FEAT;
  int t = (idx / N_FEAT) % T_SEQ;
  int n = idx / (N_FEAT * T_SEQ);
  xs[idx] = x[((long)t * N_NODES + n) * N_FEAT + f];
}

__global__ void fill_kernel(float* __restrict__ p, float v, int n) {
  int i = blockIdx.x * blockDim.x + threadIdx.x;
  if (i < n) p[i] = v;
}

__global__ void deg_count_kernel(const long long* __restrict__ ei,
                                 float* __restrict__ deg) {
  int e = blockIdx.x * blockDim.x + threadIdx.x;
  if (e >= N_EDGES) return;
  int c = (int)ei[(long)N_EDGES + e];          // col = edge_index[1]
  atomicAdd(&deg[c], 1.0f);
}

__global__ void rsqrt_kernel(float* __restrict__ p, int n) {
  int i = blockIdx.x * blockDim.x + threadIdx.x;
  if (i < n) p[i] = rsqrtf(fmaxf(p[i], 1.0f)); // deg includes self-loop
}

// self-loop contribution initializes the aggregation buffer (no atomics)
__global__ void self_init_kernel(const float* __restrict__ pre,
                                 const float* __restrict__ dis,
                                 float* __restrict__ agg) {
  int idx = blockIdx.x * blockDim.x + threadIdx.x;
  if (idx >= NT * HID) return;
  int n = idx / (T_SEQ * HID);
  float d = dis[n];
  agg[idx] = pre[idx] * d * d;
}

// one thread per (edge, feature); loops over the 12 timesteps
__global__ void scatter_kernel(const long long* __restrict__ ei,
                               const float* __restrict__ pre,
                               const float* __restrict__ dis,
                               float* __restrict__ agg) {
  int idx = blockIdx.x * blockDim.x + threadIdx.x;
  if (idx >= N_EDGES * HID) return;
  int e = idx >> 6;
  int j = idx & 63;
  int r = (int)ei[e];
  int c = (int)ei[(long)N_EDGES + e];
  float nm = dis[r] * dis[c];
  const float* src = pre + (long)r * T_SEQ * HID + j;
  float*       dst = agg + (long)c * T_SEQ * HID + j;
#pragma unroll
  for (int t = 0; t < T_SEQ; ++t)
    atomicAdd(dst + t * HID, src[t * HID] * nm);   // L2-resident f32 atomics
}

__global__ void bias_relu_kernel(float* __restrict__ h,
                                 const float* __restrict__ b) {
  int idx = blockIdx.x * blockDim.x + threadIdx.x;
  if (idx >= NT * HID) return;
  h[idx] = fmaxf(h[idx] + b[idx & 63], 0.0f);
}

__device__ __forceinline__ float sigf(float x) {
  return 1.0f / (1.0f + __expf(-x));
}

// gates g4 = [x_t ; h] @ [wih ; whh]^T  (from the fused dual-K GEMM)
__global__ void lstm_cell_kernel(const float* __restrict__ g4,
                                 const float* __restrict__ bih,
                                 const float* __restrict__ bhh,
                                 float* __restrict__ h,
                                 float* __restrict__ c,
                                 float* __restrict__ seq, int t) {
  int idx = blockIdx.x * blockDim.x + threadIdx.x;
  if (idx >= N_NODES * HID) return;
  int n = idx >> 6;
  int j = idx & 63;
  long gb = (long)n * 4 * HID;
  float gi = g4[gb + j]            + bih[j]            + bhh[j];
  float gf = g4[gb + HID + j]      + bih[HID + j]      + bhh[HID + j];
  float gg = g4[gb + 2 * HID + j]  + bih[2 * HID + j]  + bhh[2 * HID + j];
  float go = g4[gb + 3 * HID + j]  + bih[3 * HID + j]  + bhh[3 * HID + j];
  float cn = sigf(gf) * c[idx] + sigf(gi) * tanhf(gg);
  float hn = sigf(go) * tanhf(cn);
  c[idx] = cn;
  h[idx] = hn;
  seq[((long)n * T_SEQ + t) * HID + j] = hn;
}

// ---------------------------------------------------------------------------
// Host-side helpers
// ---------------------------------------------------------------------------
static inline int cdiv(long a, int b) { return (int)((a + b - 1) / b); }

template<int KT, int NCOLS, bool BT, bool DUAL, bool BIAS, bool RELU>
static void launch_gemm(const float* A0, int lda0, const float* A1, int lda1,
                        const float* B0, int ldb0, const float* B1, int ldb1,
                        float* C, int ldc, const float* bias,
                        int M, int N, hipStream_t s) {
  dim3 grid(cdiv(M >> 4, 8), N / (NCOLS * 16));
  wmma_gemm_lds<KT, NCOLS, BT, DUAL, BIAS, RELU>
      <<<grid, 256, 0, s>>>(A0, lda0, A1, lda1, B0, ldb0, B1, ldb1,
                            C, ldc, bias, M);
}

static void run_lstm(const float* in_seq, const float* wih, const float* whh,
                     const float* bih, const float* bhh,
                     float* gtmp, float* h, float* c, float* out_seq,
                     hipStream_t s) {
  fill_kernel<<<cdiv(N_NODES * HID, 256), 256, 0, s>>>(h, 0.0f, N_NODES * HID);
  fill_kernel<<<cdiv(N_NODES * HID, 256), 256, 0, s>>>(c, 0.0f, N_NODES * HID);
  for (int t = 0; t < T_SEQ; ++t) {
    // gates = [x_t ; h] @ [wih ; whh]^T   (single K=128 dual-source GEMM)
    launch_gemm<128, 4, true, true, false, false>(
        in_seq + t * HID, T_SEQ * HID,   // A0: sequence slice (k < 64)
        h, HID,                          // A1: recurrent state (k >= 64)
        wih, HID, whh, HID,              // B0 / B1 (both (256,64) row-major)
        gtmp, 4 * HID, nullptr, N_NODES, 4 * HID, s);
    lstm_cell_kernel<<<cdiv(N_NODES * HID, 256), 256, 0, s>>>(
        gtmp, bih, bhh, h, c, out_seq, t);
  }
}

// ---------------------------------------------------------------------------
// Entry point
// ---------------------------------------------------------------------------
extern "C" void kernel_launch(void* const* d_in, const int* in_sizes, int n_in,
                              void* d_out, int out_size, void* d_ws,
                              size_t ws_size, hipStream_t stream) {
  (void)in_sizes; (void)n_in; (void)out_size;
  const float*     x    = (const float*)d_in[0];
  const long long* ei   = (const long long*)d_in[1];   // int64 edge_index (2,E)
  const float* gw0  = (const float*)d_in[2];
  const float* gb0  = (const float*)d_in[3];
  const float* gw1  = (const float*)d_in[4];
  const float* gb1  = (const float*)d_in[5];
  const float* wih0 = (const float*)d_in[6];
  const float* whh0 = (const float*)d_in[7];
  const float* bih0 = (const float*)d_in[8];
  const float* bhh0 = (const float*)d_in[9];
  const float* wih1 = (const float*)d_in[10];
  const float* whh1 = (const float*)d_in[11];
  const float* bih1 = (const float*)d_in[12];
  const float* bhh1 = (const float*)d_in[13];
  const float* fc_w = (const float*)d_in[14];
  const float* fc_b = (const float*)d_in[15];
  float* out = (float*)d_out;

  // Workspace layout (floats)
  const long BIG = (long)NT * HID;              // 7,680,000
  float* ws  = (float*)d_ws;
  float* dis = ws;                 long off = 16384;
  float* B0  = ws + off;           off += BIG;
  float* B1  = ws + off;           off += BIG;
  float* B2  = ws + off;           off += BIG;
  float* GT  = ws + off;           off += (long)N_NODES * 4 * HID;
  float* Hb  = ws + off;           off += (long)N_NODES * HID;
  float* Cb  = ws + off;           off += (long)N_NODES * HID;
  if (ws_size < (size_t)off * sizeof(float)) return;

  // ---- degree / normalization -------------------------------------------
  fill_kernel<<<cdiv(N_NODES, 256), 256, 0, stream>>>(dis, 1.0f, N_NODES);
  deg_count_kernel<<<cdiv(N_EDGES, 256), 256, 0, stream>>>(ei, dis);
  rsqrt_kernel<<<cdiv(N_NODES, 256), 256, 0, stream>>>(dis, N_NODES);

  // ---- (B,T,N,F) -> (N*T,F) ---------------------------------------------
  transpose_x_kernel<<<cdiv((long)NT * N_FEAT, 256), 256, 0, stream>>>(x, B0);

  // ---- GCN layer 0: pre = xs @ gw0 ; aggregate ; relu(+bias) ------------
  launch_gemm<32, 4, false, false, false, false>(
      B0, N_FEAT, nullptr, 0, gw0, HID, nullptr, 0,
      B1, HID, nullptr, NT, HID, stream);
  self_init_kernel<<<cdiv((long)NT * HID, 256), 256, 0, stream>>>(B1, dis, B2);
  scatter_kernel<<<cdiv((long)N_EDGES * HID, 256), 256, 0, stream>>>(ei, B1, dis, B2);
  bias_relu_kernel<<<cdiv((long)NT * HID, 256), 256, 0, stream>>>(B2, gb0);

  // ---- GCN layer 1 -------------------------------------------------------
  launch_gemm<64, 4, false, false, false, false>(
      B2, HID, nullptr, 0, gw1, HID, nullptr, 0,
      B0, HID, nullptr, NT, HID, stream);
  self_init_kernel<<<cdiv((long)NT * HID, 256), 256, 0, stream>>>(B0, dis, B1);
  scatter_kernel<<<cdiv((long)N_EDGES * HID, 256), 256, 0, stream>>>(ei, B0, dis, B1);
  bias_relu_kernel<<<cdiv((long)NT * HID, 256), 256, 0, stream>>>(B1, gb1);

  // ---- LSTM stack (fused dual-K GEMM per step) --------------------------
  run_lstm(B1, wih0, whh0, bih0, bhh0, GT, Hb, Cb, B2, stream);
  run_lstm(B2, wih1, whh1, bih1, bhh1, GT, Hb, Cb, B0, stream);

  // ---- FC on last timestep: out = h[:, T-1, :] @ fc_w^T + fc_b ----------
  launch_gemm<64, 1, true, false, true, false>(
      B0 + (T_SEQ - 1) * HID, T_SEQ * HID, nullptr, 0,
      fc_w, HID, nullptr, 0, out, OUTF, fc_b, N_NODES, OUTF, stream);
}
